// TimeSeriesRNNRepresentation_74062416053120
// MI455X (gfx1250) — matrix operand order repaired and checked by reading, not compiled, not run on hardware
//
#include <hip/hip_runtime.h>
#include <math.h>

typedef __attribute__((ext_vector_type(16))) _Float16 v16h;
typedef __attribute__((ext_vector_type(8)))  float    v8f;

#define WMMA16x16x32(a, b, c) \
  __builtin_amdgcn_wmma_f32_16x16x32_f16(false, (a), false, (b), (short)0, (c), false, false)

#if __has_builtin(__builtin_amdgcn_sched_barrier)
#define SCHED_FENCE() __builtin_amdgcn_sched_barrier(0)
#else
#define SCHED_FENCE()
#endif

namespace {
constexpr int Bsz = 256;   // batch
constexpr int T   = 2048;  // time steps
constexpr int I   = 32;    // input features (== WMMA K, one k-step)
constexpr int H   = 128;   // hidden (8 N-tiles of 16, 4 K-blocks of 32)
constexpr int P   = 64;    // projection
constexpr int BT  = 16;    // batch tile (== WMMA M)
constexpr float EPS = 1e-5f;
}

// Hardware tanh (CDNA5 TRANS32 op, single v_tanh_f32).
__device__ __forceinline__ float fast_tanh(float v) {
#if __has_builtin(__builtin_amdgcn_tanhf)
  return __builtin_amdgcn_tanhf(v);
#elif __has_builtin(__builtin_amdgcn_tanh_f32)
  return __builtin_amdgcn_tanh_f32(v);
#else
  float r;
  asm volatile("v_tanh_f32 %0, %1\n\tv_nop" : "=v"(r) : "v"(v));
  return r;
#endif
}

// Only 16 workgroups exist (128 waves chip-wide): occupancy is irrelevant,
// so free the register allocator from its pressure target.
__global__ __launch_bounds__(256, 1)
void rnn2_scan_fused(const float* __restrict__ x,
                     const float* __restrict__ Wih1, const float* __restrict__ Whh1,
                     const float* __restrict__ bih1, const float* __restrict__ bhh1,
                     const float* __restrict__ Wih2, const float* __restrict__ Whh2,
                     const float* __restrict__ bih2, const float* __restrict__ bhh2,
                     const float* __restrict__ Wp,   const float* __restrict__ bp,
                     const float* __restrict__ gamma, const float* __restrict__ beta,
                     float* __restrict__ out)
{
  __shared__ _Float16 h1s[2][BT][H];   // 8 KB  (f16 WMMA operands, double buffered)
  __shared__ _Float16 h2s[2][BT][H];   // 8 KB
  __shared__ float    zbuf[BT][P];     // 4 KB  (projection staging)

  const int tid  = threadIdx.x;
  const int lane = tid & 31;
  const int wave = tid >> 5;          // 0..7 : owns hidden column tile n0..n0+15
  const int n0   = wave * 16;
  const int b0   = blockIdx.x * BT;   // batch tile origin

  // ---- zero initial hidden state (slot 0) ----
  for (int i = tid; i < BT * H; i += 256) {
    (&h1s[0][0][0])[i] = (_Float16)0.f;
    (&h2s[0][0][0])[i] = (_Float16)0.f;
  }
  __syncthreads();

  // ---- per-lane fragment coordinates (ISA 7.12.2 layouts, wave32) ----
  const int am    = lane & 15;              // A: matrix row  M = lane%16
  const int ah    = (lane >> 4) & 1;        // A: K-half select
  const int nn    = n0 + (lane & 15);       // B/C/D: column N
  const int khalf = ((lane >> 4) & 1) * 16; // B: K-half select

  // B-fragment loader: W row-major [N_total][ldk]; two 8-float chunks/lane (k = 8h.., 16+8h..).
  auto loadB = [&](const float* __restrict__ W, int ldk, int kb) {
    v16h b;
    const float* p = W + (size_t)nn * ldk + kb * 32 + (khalf >> 1);
#pragma unroll
    for (int v = 0; v < 8; ++v) {
      const int q = 2 * v + (v >> 2) * 8;
      b[2 * v]     = (_Float16)p[q];
      b[2 * v + 1] = (_Float16)p[q + 1];
    }
    return b;
  };

  // A-fragment loader from an LDS [BT][H] f16 buffer, K-block kb.
  // Two contiguous 16B chunks per lane -> ds_load_b128 x2.
  auto loadA = [&](const _Float16* __restrict__ hbuf, int kb) {
    v16h a;
    const _Float16* r = hbuf + am * H + kb * 32 + ah * 8;
#pragma unroll
    for (int v = 0; v < 8; ++v) {
      const int k = (v & 3) * 2 + (v >> 2) * 16;
      a[2 * v]     = r[k];
      a[2 * v + 1] = r[k + 1];
    }
    return a;
  };

  // A-fragment loader for x[b0.., t, :]  (I == 32 == K, single k-step).
  auto loadAx = [&](int t) {
    v16h a;
    const float* r = x + ((size_t)(b0 + am) * T + t) * I + ah * 8;
#pragma unroll
    for (int v = 0; v < 8; ++v) {
      const int k = (v & 3) * 2 + (v >> 2) * 16;
      a[2 * v]     = (_Float16)r[k];
      a[2 * v + 1] = (_Float16)r[k + 1];
    }
    return a;
  };

  // D-tile [16x16] -> hardware tanh -> f16 LDS store, column slice [n0,n0+16).
  auto storeTanh = [&](const v8f& acc, _Float16* __restrict__ dst) {
    const int nc = n0 + (lane & 15);
    const int mb = ((lane >> 4) & 1) * 8;
#pragma unroll
    for (int v = 0; v < 8; ++v)
      dst[(mb + v) * H + nc] = (_Float16)fast_tanh(acc[v]);
  };

  // ---- loop-invariant weight fragments, resident in VGPRs for all T steps ----
  v16h Bx = loadB(Wih1, I, 0);           // [128 x 32]   (layer-1 input proj)
  v16h B11[4], B21[4], B22[4];
#pragma unroll
  for (int kb = 0; kb < 4; ++kb) {
    B11[kb] = loadB(Whh1, H, kb);        // [128 x 128]  (layer-1 recurrence)
    B21[kb] = loadB(Wih2, H, kb);        // [128 x 128]  (layer-2 input proj)
    B22[kb] = loadB(Whh2, H, kb);        // [128 x 128]  (layer-2 recurrence)
  }
  const float bias1 = bih1[nn] + bhh1[nn];
  const float bias2 = bih2[nn] + bhh2[nn];

  // ---- sequential scan over T: one barrier per step.
  // x_t fragment is software-pipelined: loads for t+1 issue during layer 2.
  v16h ax = loadAx(0);
  for (int t = 0; t < T; ++t) {
    const int cur = t & 1, nxt = cur ^ 1;

    if (wave == 0 && t + 8 < T)
      __builtin_prefetch(x + ((size_t)(b0 + am) * T + (t + 8)) * I + ah * 8, 0, 0);

    // ---- layer 1: h1' = tanh(x_t Wih1^T + h1 Whh1^T + bias1) ----
    // Issue ALL A-fragment loads, then fence the scheduler so they stay
    // grouped ahead of the WMMA chain: latencies overlap, waits are staged
    // (dscnt<=6,4,2,0) instead of four serial dscnt==0 round-trips.
    const _Float16* h1c = &h1s[cur][0][0];
    v16h a0 = loadA(h1c, 0);
    v16h a1 = loadA(h1c, 1);
    v16h a2 = loadA(h1c, 2);
    v16h a3 = loadA(h1c, 3);
    SCHED_FENCE();
    v8f acc1 = {bias1, bias1, bias1, bias1, bias1, bias1, bias1, bias1};
    acc1 = WMMA16x16x32(ax, Bx,     acc1);
    acc1 = WMMA16x16x32(a0, B11[0], acc1);
    acc1 = WMMA16x16x32(a1, B11[1], acc1);
    acc1 = WMMA16x16x32(a2, B11[2], acc1);
    acc1 = WMMA16x16x32(a3, B11[3], acc1);
    storeTanh(acc1, &h1s[nxt][0][0]);
    __syncthreads();   // h1' visible to all waves for layer 2

    // ---- layer 2: h2' = tanh(h1' Wih2^T + h2 Whh2^T + bias2) ----
    const _Float16* h1n = &h1s[nxt][0][0];
    const _Float16* h2c = &h2s[cur][0][0];
    v16h c0 = loadA(h1n, 0);
    v16h c1 = loadA(h1n, 1);
    v16h c2 = loadA(h1n, 2);
    v16h c3 = loadA(h1n, 3);
    v16h d0 = loadA(h2c, 0);
    v16h d1 = loadA(h2c, 1);
    v16h d2 = loadA(h2c, 2);
    v16h d3 = loadA(h2c, 3);
    // Next-step x loads issue here too; global latency hides under 8 WMMAs.
    const int tn = (t + 1 < T) ? (t + 1) : t;
    v16h axn = loadAx(tn);
    SCHED_FENCE();

    v8f acc2 = {bias2, bias2, bias2, bias2, bias2, bias2, bias2, bias2};
    acc2 = WMMA16x16x32(c0, B21[0], acc2);
    acc2 = WMMA16x16x32(c1, B21[1], acc2);
    acc2 = WMMA16x16x32(c2, B21[2], acc2);
    acc2 = WMMA16x16x32(c3, B21[3], acc2);
    acc2 = WMMA16x16x32(d0, B22[0], acc2);
    acc2 = WMMA16x16x32(d1, B22[1], acc2);
    acc2 = WMMA16x16x32(d2, B22[2], acc2);
    acc2 = WMMA16x16x32(d3, B22[3], acc2);
    storeTanh(acc2, &h2s[nxt][0][0]);
    ax = axn;
    // no end-of-step barrier: next step's mid barrier covers the hazards
  }
  __syncthreads();     // last h2 write visible before projection

  const int fin = ((T - 1) & 1) ^ 1;   // slot holding h2[T-1]

  // ---- projection z = h2 Wp^T + bp  (P=64 -> waves 0..3, wave-uniform) ----
  if (wave < 4) {
    v16h BW[4];
#pragma unroll
    for (int kb = 0; kb < 4; ++kb) BW[kb] = loadB(Wp, H, kb);  // nn in [0,64)
    const float bpv = bp[nn];
    v8f accp = {bpv, bpv, bpv, bpv, bpv, bpv, bpv, bpv};
#pragma unroll
    for (int kb = 0; kb < 4; ++kb)
      accp = WMMA16x16x32(loadA(&h2s[fin][0][0], kb), BW[kb], accp);
    const int mb = ((lane >> 4) & 1) * 8;
#pragma unroll
    for (int v = 0; v < 8; ++v) zbuf[mb + v][nn] = accp[v];
  }
  __syncthreads();

  // ---- layernorm over P=64 + tanh, one batch row per thread ----
  if (tid < BT) {
    float mu = 0.f;
#pragma unroll
    for (int p = 0; p < P; ++p) mu += zbuf[tid][p];
    mu *= (1.f / P);
    float var = 0.f;
#pragma unroll
    for (int p = 0; p < P; ++p) {
      const float d = zbuf[tid][p] - mu;
      var += d * d;
    }
    var *= (1.f / P);
    const float inv = rsqrtf(var + EPS);
#pragma unroll
    for (int p = 0; p < P; ++p)
      out[(size_t)(b0 + tid) * P + p] =
          fast_tanh((zbuf[tid][p] - mu) * inv * gamma[p] + beta[p]);
  }
}

extern "C" void kernel_launch(void* const* d_in, const int* in_sizes, int n_in,
                              void* d_out, int out_size, void* d_ws, size_t ws_size,
                              hipStream_t stream) {
  (void)in_sizes; (void)n_in; (void)out_size; (void)d_ws; (void)ws_size;
  const float* xp    = (const float*)d_in[0];
  const float* Wih1  = (const float*)d_in[1];
  const float* Whh1  = (const float*)d_in[2];
  const float* bih1  = (const float*)d_in[3];
  const float* bhh1  = (const float*)d_in[4];
  const float* Wih2  = (const float*)d_in[5];
  const float* Whh2  = (const float*)d_in[6];
  const float* bih2  = (const float*)d_in[7];
  const float* bhh2  = (const float*)d_in[8];
  const float* Wp    = (const float*)d_in[9];
  const float* bp    = (const float*)d_in[10];
  const float* gamma = (const float*)d_in[11];
  const float* beta  = (const float*)d_in[12];
  float* out = (float*)d_out;

  dim3 grid(Bsz / BT);   // 16 batch tiles
  dim3 block(256);       // 8 wave32 waves: one hidden column tile each
  rnn2_scan_fused<<<grid, block, 0, stream>>>(xp, Wih1, Whh1, bih1, bhh1,
                                              Wih2, Whh2, bih2, bhh2,
                                              Wp, bp, gamma, beta, out);
}
